// VQVAE_4561255269151
// MI455X (gfx1250) — matrix operand order repaired and checked by reading, not compiled, not run on hardware
//
#include <hip/hip_runtime.h>
#include <cstddef>

typedef __attribute__((ext_vector_type(16))) __bf16 v16bf;
typedef __attribute__((ext_vector_type(8)))  float  v8f;
typedef unsigned int   u32;
typedef unsigned short u16;

union Frag16 { v16bf v; u32 u[8]; };

// native hardware f32 -> bf16 convert (RNE), returned as raw bits
__device__ __forceinline__ u16 f2bf(float f) {
  return __builtin_bit_cast(u16, (__bf16)f);
}

__device__ __forceinline__ float gelu_erf(float x) {
  return 0.5f * x * (1.0f + erff(x * 0.70710678118654752f));
}

// ---------------------------------------------------------------------------
// Packing / conversion kernels
// ---------------------------------------------------------------------------
__global__ void k_packWT(const float* __restrict__ w, u16* __restrict__ o,
                         int din, int dout) {
  int n = din * dout;
  for (int id = blockIdx.x * blockDim.x + threadIdx.x; id < n;
       id += gridDim.x * blockDim.x) {
    int j = id / din, i = id - j * din;                        // o[j][i] = w[i][j]
    o[id] = f2bf(w[(size_t)i * dout + j]);
  }
}

__global__ void k_cvt(const float* __restrict__ in, u16* __restrict__ o, int n) {
  for (int id = blockIdx.x * blockDim.x + threadIdx.x; id < n;
       id += gridDim.x * blockDim.x)
    o[id] = f2bf(in[id]);
}

__global__ void k_rownorm(const float* __restrict__ cb, float* __restrict__ cn) {
  int c = blockIdx.x * blockDim.x + threadIdx.x;
  if (c < 1024) {
    float s = 0.f;
    for (int j = 0; j < 128; ++j) { float v = cb[c * 128 + j]; s += v * v; }
    cn[c] = s;
  }
}

// ---------------------------------------------------------------------------
// Fused GEMM (+bias [+LayerNorm] [+GELU]) kernel.
//   A: [32768][K] bf16 row-major, W: [N][K] bf16 (pre-transposed weights)
//   Block: WAVES waves, owns 64 rows x full N (waves split N).
//   All B fragments for a k-chunk are loaded before any WMMA so the
//   scheduler can emit one load clause and overlap waits with the MMA pipe.
// ---------------------------------------------------------------------------
template <int K, int N, int WAVES, bool LN, bool GELU_ON, bool WF32, bool WBF>
__global__ __launch_bounds__(WAVES * 32) void k_gemm(
    const u16* __restrict__ A, const u16* __restrict__ W,
    const float* __restrict__ bias, const float* __restrict__ gam,
    const float* __restrict__ bet, float* __restrict__ outF,
    u16* __restrict__ outB) {
  constexpr int BLOCK = WAVES * 32;
  constexpr int TPW   = N / (16 * WAVES);   // 16-col tiles per wave
  constexpr int KU    = K / 2;              // A row stride in dwords
  constexpr int NPW   = N / WAVES;

  __shared__ u32    sA[64 * 17];            // 64x32 bf16 A tile, stride 17 dwords
  __shared__ float2 sWS[WAVES][64];         // per-wave (sum, sumsq) per row
  __shared__ float2 sRS[64];                // per-row (mean, rstd)

  const int tid  = threadIdx.x;
  const int wv   = tid >> 5;
  const int lane = tid & 31;
  const int l16  = lane & 15;
  const int half = lane >> 4;
  const int row0 = blockIdx.x * 64;
  const int nw   = wv * NPW;

  const u32* Ag = (const u32*)A;

  const u32* bp[TPW];
#pragma unroll
  for (int t = 0; t < TPW; ++t) {
    int n = nw + t * 16 + l16;
    bp[t] = (const u32*)W + (size_t)n * KU + 8 * half;
  }

  v8f zero = {};
  v8f acc[4][TPW];
#pragma unroll
  for (int rt = 0; rt < 4; ++rt)
#pragma unroll
    for (int t = 0; t < TPW; ++t) acc[rt][t] = zero;

  for (int kc = 0; kc < K; kc += 32) {
    const int ku = kc >> 1;
    __syncthreads();
#pragma unroll
    for (int i = 0; i < 1024 / BLOCK; ++i) {  // stage 64x32 bf16 A tile
      int idx = tid + BLOCK * i;
      int r = idx >> 4, p = idx & 15;
      sA[r * 17 + p] = Ag[(size_t)(row0 + r) * KU + ku + p];
    }
    __syncthreads();

    // ---- load ALL B fragments for this chunk (one big load clause) ----
    Frag16 bf[TPW];
#pragma unroll
    for (int t = 0; t < TPW; ++t) {           // B layout: k = e + 16*half, contiguous
      const uint4* q = (const uint4*)(bp[t] + ku);
      uint4 b0 = q[0], b1 = q[1];
      bf[t].u[0] = b0.x; bf[t].u[1] = b0.y; bf[t].u[2] = b0.z; bf[t].u[3] = b0.w;
      bf[t].u[4] = b1.x; bf[t].u[5] = b1.y; bf[t].u[6] = b1.z; bf[t].u[7] = b1.w;
      if (kc + 32 < K) __builtin_prefetch(bp[t] + ku + 16, 0, 1);
    }

    // ---- A fragments from LDS (bank-conflict-free, stride 17) ----
    Frag16 af[4];
#pragma unroll
    for (int rt = 0; rt < 4; ++rt) {          // A layout: K 0..15 (+16*half), contiguous
      int base = (rt * 16 + l16) * 17 + 8 * half;
#pragma unroll
      for (int v = 0; v < 4; ++v) {
        af[rt].u[v]     = sA[base + v];
        af[rt].u[v + 4] = sA[base + 4 + v];
      }
    }

    // ---- 4*TPW WMMAs back-to-back ----
#pragma unroll
    for (int t = 0; t < TPW; ++t)
#pragma unroll
      for (int rt = 0; rt < 4; ++rt)
        acc[rt][t] = __builtin_amdgcn_wmma_f32_16x16x32_bf16(
            false, af[rt].v, false, bf[t].v, (short)0, acc[rt][t], false, false);
  }

  // ---------------- epilogue: bias (+LN) (+GELU) ----------------
  float bv[TPW], gv[TPW], ev[TPW];
#pragma unroll
  for (int t = 0; t < TPW; ++t) {
    int n = nw + t * 16 + l16;
    bv[t] = bias[n];
    if constexpr (LN) { gv[t] = gam[n]; ev[t] = bet[n]; }
  }
#pragma unroll
  for (int rt = 0; rt < 4; ++rt)
#pragma unroll
    for (int t = 0; t < TPW; ++t)
#pragma unroll
      for (int r = 0; r < 8; ++r) acc[rt][t][r] += bv[t];

  if constexpr (LN) {
#pragma unroll
    for (int rt = 0; rt < 4; ++rt)
#pragma unroll
      for (int r = 0; r < 8; ++r) {
        float sx = 0.f, sq = 0.f;
#pragma unroll
        for (int t = 0; t < TPW; ++t) { float x = acc[rt][t][r]; sx += x; sq += x * x; }
#pragma unroll
        for (int m = 1; m < 16; m <<= 1) {    // reduce over 16 cols (same half)
          sx += __shfl_xor(sx, m, 32);
          sq += __shfl_xor(sq, m, 32);
        }
        if (l16 == 0) sWS[wv][rt * 16 + 8 * half + r] = make_float2(sx, sq);
      }
    __syncthreads();
    if (tid < 64) {
      float sx = 0.f, sq = 0.f;
#pragma unroll
      for (int w = 0; w < WAVES; ++w) { float2 p2 = sWS[w][tid]; sx += p2.x; sq += p2.y; }
      float mu  = sx / (float)N;
      float var = sq / (float)N - mu * mu;
      sRS[tid] = make_float2(mu, rsqrtf(var + 1e-5f));
    }
    __syncthreads();
  }

#pragma unroll
  for (int rt = 0; rt < 4; ++rt)
#pragma unroll
    for (int r = 0; r < 8; ++r) {
      const int ri = rt * 16 + 8 * half + r;
      const size_t row = (size_t)(row0 + ri);
      float mu = 0.f, rs = 1.f;
      if constexpr (LN) { float2 s = sRS[ri]; mu = s.x; rs = s.y; }
#pragma unroll
      for (int t = 0; t < TPW; ++t) {
        int n = nw + t * 16 + l16;
        float x = acc[rt][t][r];
        if constexpr (LN) x = (x - mu) * rs * gv[t] + ev[t];
        if constexpr (GELU_ON) x = gelu_erf(x);
        if constexpr (WF32) outF[row * N + n] = x;
        if constexpr (WBF)  outB[row * N + n] = f2bf(x);
      }
    }
}

// ---------------------------------------------------------------------------
// VQ: nearest codebook entry per token (argmin over 1024 codes, E=128 via WMMA)
// block = 16 tokens x all 1024 codes (8 waves x 128 codes each)
// ---------------------------------------------------------------------------
__global__ __launch_bounds__(256) void k_vq(
    const float* __restrict__ E, const u16* __restrict__ Cb,
    const float* __restrict__ Cf, const float* __restrict__ cn,
    float* __restrict__ idxOut, u16* __restrict__ Q,
    float* __restrict__ part) {
  __shared__ u32   sA[16 * 17];
  __shared__ float sMD[8][16];
  __shared__ int   sMI[8][16];
  __shared__ int   sSel[16];
  __shared__ float sRed[256];

  const int tid = threadIdx.x;
  const int wv = tid >> 5, lane = tid & 31, l16 = lane & 15, half = lane >> 4;
  const int row0 = blockIdx.x * 16;

  const u32* bp[8];
#pragma unroll
  for (int t = 0; t < 8; ++t) {
    int n = wv * 128 + t * 16 + l16;
    bp[t] = (const u32*)Cb + (size_t)n * 64 + 8 * half;
  }

  v8f zero = {};
  v8f acc[8];
#pragma unroll
  for (int t = 0; t < 8; ++t) acc[t] = zero;

  const int rr = tid >> 4, pp = tid & 15;
  for (int kc = 0; kc < 128; kc += 32) {
    __syncthreads();
    float2 v2 = *(const float2*)(E + (size_t)(row0 + rr) * 128 + kc + 2 * pp);
    sA[rr * 17 + pp] = (u32)f2bf(v2.x) | ((u32)f2bf(v2.y) << 16);
    __syncthreads();

    Frag16 bf[8];
#pragma unroll
    for (int t = 0; t < 8; ++t) {
      const uint4* q = (const uint4*)(bp[t] + (kc >> 1));
      uint4 b0 = q[0], b1 = q[1];
      bf[t].u[0] = b0.x; bf[t].u[1] = b0.y; bf[t].u[2] = b0.z; bf[t].u[3] = b0.w;
      bf[t].u[4] = b1.x; bf[t].u[5] = b1.y; bf[t].u[6] = b1.z; bf[t].u[7] = b1.w;
    }
    Frag16 af;
    int base = l16 * 17 + 8 * half;
#pragma unroll
    for (int v = 0; v < 4; ++v) { af.u[v] = sA[base + v]; af.u[v + 4] = sA[base + 4 + v]; }
#pragma unroll
    for (int t = 0; t < 8; ++t)
      acc[t] = __builtin_amdgcn_wmma_f32_16x16x32_bf16(
          false, af.v, false, bf[t].v, (short)0, acc[t], false, false);
  }

  float cnv[8];
#pragma unroll
  for (int t = 0; t < 8; ++t) cnv[t] = cn[wv * 128 + t * 16 + l16];

#pragma unroll
  for (int r = 0; r < 8; ++r) {                 // argmin (||e||^2 is row-const, dropped)
    float best = 3.4e38f; int bi = 0x7fffffff;
#pragma unroll
    for (int t = 0; t < 8; ++t) {
      float d = cnv[t] - 2.0f * acc[t][r];
      int n = wv * 128 + t * 16 + l16;
      if (d < best || (d == best && n < bi)) { best = d; bi = n; }
    }
#pragma unroll
    for (int m = 1; m < 16; m <<= 1) {
      float ob = __shfl_xor(best, m, 32);
      int   oi = __shfl_xor(bi, m, 32);
      if (ob < best || (ob == best && oi < bi)) { best = ob; bi = oi; }
    }
    if (l16 == 0) { sMD[wv][8 * half + r] = best; sMI[wv][8 * half + r] = bi; }
  }
  __syncthreads();
  if (tid < 16) {
    float best = 3.4e38f; int bi = 0x7fffffff;
    for (int w = 0; w < 8; ++w) {
      float b = sMD[w][tid]; int i = sMI[w][tid];
      if (b < best || (b == best && i < bi)) { best = b; bi = i; }
    }
    sSel[tid] = bi;
    idxOut[row0 + tid] = (float)bi;
  }
  __syncthreads();

  {                                             // gather q (bf16) + commit partial
    const int tok = tid >> 4, p = tid & 15;
    const int sel = sSel[tok];
    float cp = 0.f;
#pragma unroll
    for (int j = 0; j < 8; ++j) {
      int e = p * 8 + j;
      float cv  = Cf[(size_t)sel * 128 + e];
      float evv = E[(size_t)(row0 + tok) * 128 + e];
      float dd = cv - evv;
      cp += dd * dd;
      Q[(size_t)(row0 + tok) * 128 + e] = f2bf(cv);
    }
    sRed[tid] = cp;
  }
  __syncthreads();
  for (int m = 128; m > 0; m >>= 1) {
    if (tid < m) sRed[tid] += sRed[tid + m];
    __syncthreads();
  }
  if (tid == 0) part[blockIdx.x] = sRed[0];
}

__global__ void k_final(const float* __restrict__ part, float* __restrict__ out) {
  __shared__ float s[256];
  float a = 0.f;
  for (int i = threadIdx.x; i < 2048; i += 256) a += part[i];
  s[threadIdx.x] = a;
  __syncthreads();
  for (int m = 128; m > 0; m >>= 1) {
    if (threadIdx.x < m) s[threadIdx.x] += s[threadIdx.x + m];
    __syncthreads();
  }
  if (threadIdx.x == 0) out[0] = s[0] * (1.0f / (32768.0f * 128.0f));
}

// ---------------------------------------------------------------------------
extern "C" void kernel_launch(void* const* d_in, const int* in_sizes, int n_in,
                              void* d_out, int out_size, void* d_ws, size_t ws_size,
                              hipStream_t stream) {
  (void)in_sizes; (void)n_in; (void)out_size; (void)ws_size;
  const int M = 32768;
  const size_t RECON = (size_t)M * 512;

  auto F = [&](int i) { return (const float*)d_in[i]; };

  char* ws = (char*)d_ws;
  size_t off = 0;
  auto carve = [&](size_t bytes) -> void* {
    void* p = ws + off;
    off = (off + bytes + 255) & ~(size_t)255;
    return p;
  };

  // region 1: ping activation buffer (bf16, M x 1024)
  u16* h1 = (u16*)carve((size_t)M * 1024 * 2);
  // region 2: pong buffer; aliases x_bf16 / enc-embedding(eF) / q by lifetime
  char* r2 = (char*)carve((size_t)M * 1024 * 2);
  u16*   h2 = (u16*)r2;
  u16*   xb = (u16*)r2;                            // enc1 input (dead after enc1)
  float* eF = (float*)r2;                          // enc4 output (fp32, M x 128)
  u16*   qB = (u16*)(r2 + (size_t)M * 128 * 4);    // VQ output (bf16, M x 128)

  const int edims[4][2] = {{512, 1024}, {1024, 1024}, {1024, 1024}, {1024, 128}};
  const int ddims[4][2] = {{128, 1024}, {1024, 1024}, {1024, 1024}, {1024, 512}};
  u16* We[4];
  u16* Wd[4];
  for (int i = 0; i < 4; ++i) We[i] = (u16*)carve((size_t)edims[i][0] * edims[i][1] * 2);
  for (int i = 0; i < 4; ++i) Wd[i] = (u16*)carve((size_t)ddims[i][0] * ddims[i][1] * 2);
  u16*   cbB  = (u16*)carve((size_t)1024 * 128 * 2);
  float* cnrm = (float*)carve(1024 * 4);
  float* part = (float*)carve(2048 * 4);

  float* outF = (float*)d_out;

  // ---- pack weights (transposed bf16), codebook, input ----
  k_cvt<<<4096, 256, 0, stream>>>(F(0), xb, M * 512);
  for (int i = 0; i < 4; ++i)
    k_packWT<<<1024, 256, 0, stream>>>(F(1 + 4 * i), We[i], edims[i][0], edims[i][1]);
  for (int i = 0; i < 4; ++i)
    k_packWT<<<1024, 256, 0, stream>>>(F(17 + 4 * i), Wd[i], ddims[i][0], ddims[i][1]);
  k_cvt<<<64, 256, 0, stream>>>(F(31), cbB, 1024 * 128);
  k_rownorm<<<4, 256, 0, stream>>>(F(31), cnrm);

  // ---- encoder (64-row M-tiles => 512 blocks) ----
  k_gemm<512, 1024, 16, true, true, false, true><<<512, 512, 0, stream>>>(
      xb, We[0], F(2), F(3), F(4), nullptr, h1);
  k_gemm<1024, 1024, 16, true, true, false, true><<<512, 512, 0, stream>>>(
      h1, We[1], F(6), F(7), F(8), nullptr, h2);
  k_gemm<1024, 1024, 16, true, true, false, true><<<512, 512, 0, stream>>>(
      h2, We[2], F(10), F(11), F(12), nullptr, h1);
  k_gemm<1024, 128, 8, true, false, true, false><<<512, 256, 0, stream>>>(
      h1, We[3], F(14), F(15), F(16), eF, nullptr);

  // ---- vector quantization ----
  k_vq<<<2048, 256, 0, stream>>>(eF, cbB, F(31), cnrm,
                                 outF + RECON + 1, qB, part);

  // ---- decoder ----
  k_gemm<128, 1024, 16, true, true, false, true><<<512, 512, 0, stream>>>(
      qB, Wd[0], F(18), F(19), F(20), nullptr, h1);
  k_gemm<1024, 1024, 16, true, true, false, true><<<512, 512, 0, stream>>>(
      h1, Wd[1], F(22), F(23), F(24), nullptr, h2);
  k_gemm<1024, 1024, 16, true, true, false, true><<<512, 512, 0, stream>>>(
      h2, Wd[2], F(26), F(28 - 1), F(28), nullptr, h1);
  k_gemm<1024, 512, 16, false, false, true, false><<<512, 512, 0, stream>>>(
      h1, Wd[3], F(30), nullptr, nullptr, outF, nullptr);

  // ---- commit loss ----
  k_final<<<1, 256, 0, stream>>>(part, outF + RECON);
}
